// MSTPFormerBlock_18476949307585
// MI455X (gfx1250) — matrix operand order repaired and checked by reading, not compiled, not run on hardware
//
#include <hip/hip_runtime.h>
#include <math.h>
#include <stdint.h>

// MI455X / gfx1250, wave32. All GEMMs via v_wmma_f32_16x16x32_f16 (f32 accum).
// Activations are L2-resident (<<192MB) -> compute-limited. WMMA fragments are
// fetched DIRECTLY from global memory with contiguous b128 loads (fragment
// layouts use k = 16h+s for A and B alike, matching row-major A[f32] and W[N,K] f16): no LDS
// in the GEMM. The sequential Mamba scan uses GLOBAL_LOAD_ASYNC_TO_LDS double
// buffering (ASYNCcnt) to hide per-step L2 latency.

typedef __attribute__((ext_vector_type(16))) _Float16 v16h;
typedef __attribute__((ext_vector_type(4)))  _Float16 v4h;
typedef __attribute__((ext_vector_type(8)))  float    v8f;
typedef __attribute__((ext_vector_type(4)))  int      v4i;

#define NTOK    8262      // B*T*J
#define B_DIM   2
#define T_DIM   243
#define J_DIM   17
#define N_HEADS 8
#define EPS_F   1e-5f

// ---------------- async global->LDS (gfx1250 ASYNCcnt path) -----------------
#if __has_builtin(__builtin_amdgcn_global_load_async_to_lds_b128) && \
    __has_builtin(__builtin_amdgcn_s_wait_asynccnt)
#define HAVE_ASYNC_LDS 1
#else
#define HAVE_ASYNC_LDS 0
#endif

#if HAVE_ASYNC_LDS
typedef __attribute__((address_space(1))) v4i gas_v4i;  // global int4*
typedef __attribute__((address_space(3))) v4i las_v4i;  // lds int4*
#endif

__device__ __forceinline__ void async_copy16(const float* gsrc, float* lds_dst) {
#if HAVE_ASYNC_LDS
  // generic->as1: global generic addr == as1 addr; generic LDS low 32 bits ==
  // LDS byte offset (aperture rule), so trunc-to-32 gives the as3 pointer.
  __builtin_amdgcn_global_load_async_to_lds_b128(
      (gas_v4i*)(uintptr_t)gsrc,
      (las_v4i*)(uint32_t)(uintptr_t)lds_dst,
      0, 0);
#else
  *(float4*)lds_dst = *(const float4*)gsrc;  // synchronous fallback
#endif
}
__device__ __forceinline__ void async_wait0() {
#if HAVE_ASYNC_LDS
  __builtin_amdgcn_s_wait_asynccnt(0);
#endif
}

// ------------------------------ helpers ------------------------------------
__device__ __forceinline__ float wave_sum(float v) {
#pragma unroll
  for (int m = 16; m >= 1; m >>= 1) v += __shfl_xor(v, m, 32);  // wave32 butterfly
  return v;
}
__device__ __forceinline__ float silu_f(float x) { return x / (1.f + __expf(-x)); }
__device__ __forceinline__ float4 ld4(const float* p) { return *(const float4*)p; }
__device__ __forceinline__ void st4(float* p, float4 v) { *(float4*)p = v; }
__device__ __forceinline__ float dot4(float4 a, float4 b) {
  return a.x * b.x + a.y * b.y + a.z * b.z + a.w * b.w;
}

template <int ACT>
__device__ __forceinline__ float act_f(float v) {
  if constexpr (ACT == 1)  // exact GELU
    return 0.5f * v * (1.f + erff(v * 0.70710678118654752f));
  return v;
}

// ------------------------- f32 -> f16 weight staging ------------------------
__global__ void f32_to_f16_kernel(const float* __restrict__ in, _Float16* __restrict__ out, int n4) {
  int i = blockIdx.x * 256 + threadIdx.x;
  if (i >= n4) return;
  float4 v = ld4(in + i * 4);
  v4h h = { (_Float16)v.x, (_Float16)v.y, (_Float16)v.z, (_Float16)v.w };
  *(v4h*)(out + i * 4) = h;
}

// ------------------------------ WMMA GEMM -----------------------------------
// C[M,N] = act(A[M,K] @ W[N,K]^T + bias). Block = 128 thr = 4 waves.
// Block tile 64x64: wave w owns rows [m0+16w, +16), 4 N-tiles of 16.
// Per K-step(32): A frag = 4x b128 f32 loads + cvt; each B frag = 32B vec load.
template <int ACT>
__global__ __launch_bounds__(128) void gemm_wmma_f16(
    const float* __restrict__ A, const _Float16* __restrict__ W,
    const float* __restrict__ bias, float* __restrict__ C, int M, int N, int K) {
  const int wave = threadIdx.x >> 5;
  const int lane = threadIdx.x & 31;
  const int half = lane >> 4;
  const int l16  = lane & 15;
  const int m0 = blockIdx.y * 64;
  const int n0 = blockIdx.x * 64;
  const int gm = m0 + (wave << 4) + l16;   // A row owned by this lane
  const bool mok = gm < M;
  const float* arow = A + (size_t)(mok ? gm : 0) * K;
  const bool n1ok = (n0 + 16 + l16) < N;
  const bool n2ok = (n0 + 32 + l16) < N;
  const bool n3ok = (n0 + 48 + l16) < N;
  const _Float16* wrow = W + (size_t)(n0 + l16) * K + (half << 4);

  v8f acc0 = {}, acc1 = {}, acc2 = {}, acc3 = {};
  for (int kb = 0; kb < K; kb += 32) {
    if (mok && kb + 32 < K) __builtin_prefetch(arow + kb + 32, 0, 0);  // global_prefetch_b8
    // ---- A fragment: lane holds K = [kb+16h, +16) of its row
    float4 a0 = {0, 0, 0, 0}, a1 = a0, a2 = a0, a3 = a0;
    if (mok) {
      const float* ap = arow + kb + (half << 4);
      a0 = ld4(ap);      a1 = ld4(ap + 4);        // K = 16h + 0..7
      a2 = ld4(ap + 8);  a3 = ld4(ap + 12);       // K = 16h + 8..15
    }
    v16h af;
    af[0] = (_Float16)a0.x;  af[1] = (_Float16)a0.y;  af[2] = (_Float16)a0.z;  af[3] = (_Float16)a0.w;
    af[4] = (_Float16)a1.x;  af[5] = (_Float16)a1.y;  af[6] = (_Float16)a1.z;  af[7] = (_Float16)a1.w;
    af[8] = (_Float16)a2.x;  af[9] = (_Float16)a2.y;  af[10] = (_Float16)a2.z; af[11] = (_Float16)a2.w;
    af[12] = (_Float16)a3.x; af[13] = (_Float16)a3.y; af[14] = (_Float16)a3.z; af[15] = (_Float16)a3.w;
    // ---- B fragments: lane holds W[n0+16i+l16][kb+16h .. +16) (contiguous!)
    v16h bf0 = {}, bf1 = {}, bf2 = {}, bf3 = {};
    {
      const _Float16* wp = wrow + kb;
      if (n0 + l16 < N) bf0 = *(const v16h*)(wp);
      if (n1ok)         bf1 = *(const v16h*)(wp + (size_t)16 * K);
      if (n2ok)         bf2 = *(const v16h*)(wp + (size_t)32 * K);
      if (n3ok)         bf3 = *(const v16h*)(wp + (size_t)48 * K);
    }
    acc0 = __builtin_amdgcn_wmma_f32_16x16x32_f16(false, af, false, bf0, (short)0, acc0, false, false);
    acc1 = __builtin_amdgcn_wmma_f32_16x16x32_f16(false, af, false, bf1, (short)0, acc1, false, false);
    acc2 = __builtin_amdgcn_wmma_f32_16x16x32_f16(false, af, false, bf2, (short)0, acc2, false, false);
    acc3 = __builtin_amdgcn_wmma_f32_16x16x32_f16(false, af, false, bf3, (short)0, acc3, false, false);
  }
  // ---- epilogue: C layout VGPR r -> M = r + 8*half, N = l16
  const int rbase = m0 + (wave << 4) + (half << 3);
#define EPILOG(ACCV, I)                                                        \
  {                                                                            \
    const int gn = n0 + ((I) << 4) + l16;                                      \
    if (gn < N) {                                                              \
      const float bv = bias ? bias[gn] : 0.f;                                  \
      _Pragma("unroll") for (int r = 0; r < 8; ++r) {                          \
        int grow = rbase + r;                                                  \
        if (grow < M) C[(size_t)grow * N + gn] = act_f<ACT>(ACCV[r] + bv);     \
      }                                                                        \
    }                                                                          \
  }
  EPILOG(acc0, 0) EPILOG(acc1, 1) EPILOG(acc2, 2) EPILOG(acc3, 3)
#undef EPILOG
}

// ------------------------------ LayerNorm -----------------------------------
__global__ __launch_bounds__(128) void ln_kernel(
    const float* __restrict__ x, const float* __restrict__ w,
    const float* __restrict__ b, float* __restrict__ y, int ntok) {
  int wv   = (blockIdx.x * blockDim.x + threadIdx.x) >> 5;
  int lane = threadIdx.x & 31;
  if (wv >= ntok) return;
  const int c = lane * 4;  // 128 ch / 32 lanes = float4 each
  float4 v = ld4(x + (size_t)wv * 128 + c);
  float s = wave_sum(v.x + v.y + v.z + v.w);
  float mean = s * (1.f / 128.f);
  float4 d = { v.x - mean, v.y - mean, v.z - mean, v.w - mean };
  float q = wave_sum(dot4(d, d));
  float rs = rsqrtf(q * (1.f / 128.f) + EPS_F);
  float4 wv4 = ld4(w + c), bv4 = ld4(b + c);
  float4 o = { d.x * rs * wv4.x + bv4.x, d.y * rs * wv4.y + bv4.y,
               d.z * rs * wv4.z + bv4.z, d.w * rs * wv4.w + bv4.w };
  st4(y + (size_t)wv * 128 + c, o);
}

// --------------------------- residual / scaled add --------------------------
__global__ void add_scaled_kernel(float* __restrict__ x, const float* __restrict__ y,
                                  const float* __restrict__ ls, int n4) {
  int i = blockIdx.x * 256 + threadIdx.x;
  if (i >= n4) return;
  int c = (i * 4) & 127;
  float4 xv = ld4(x + i * 4), yv = ld4(y + i * 4);
  float4 s = ls ? ld4(ls + c) : float4{1.f, 1.f, 1.f, 1.f};
  xv.x += s.x * yv.x; xv.y += s.y * yv.y; xv.z += s.z * yv.z; xv.w += s.w * yv.w;
  st4(x + i * 4, xv);
}

// ------------------------------ attention -----------------------------------
// qkv buffer: [tok, 384] with q=[0,128), k=[128,256), v=[256,384); head h -> h*16+d
__global__ void attn_spatial_kernel(const float* __restrict__ qkv, float* __restrict__ o) {
  int idx = blockIdx.x * blockDim.x + threadIdx.x;
  if (idx >= B_DIM * T_DIM * N_HEADS * J_DIM) return;
  int q  = idx % J_DIM;
  int h  = (idx / J_DIM) % N_HEADS;
  int bt = idx / (J_DIM * N_HEADS);
  const float scale = 0.25f;  // 16^-0.5
  size_t base = (size_t)bt * J_DIM * 384 + h * 16;
  const float* qp = qkv + base + q * 384;
  float4 q0 = ld4(qp), q1 = ld4(qp + 4), q2 = ld4(qp + 8), q3 = ld4(qp + 12);
  float sc[J_DIM];
  float mx = -1e30f;
  for (int k = 0; k < J_DIM; ++k) {
    const float* kp = qkv + base + k * 384 + 128;
    float s = dot4(q0, ld4(kp)) + dot4(q1, ld4(kp + 4)) +
              dot4(q2, ld4(kp + 8)) + dot4(q3, ld4(kp + 12));
    s *= scale; sc[k] = s; mx = fmaxf(mx, s);
  }
  float l = 0.f;
  for (int k = 0; k < J_DIM; ++k) { float e = __expf(sc[k] - mx); sc[k] = e; l += e; }
  float4 a0 = {}, a1 = {}, a2 = {}, a3 = {};
  for (int k = 0; k < J_DIM; ++k) {
    const float* vp = qkv + base + k * 384 + 256;
    float p = sc[k];
    float4 v0 = ld4(vp), v1 = ld4(vp + 4), v2 = ld4(vp + 8), v3 = ld4(vp + 12);
    a0.x += p * v0.x; a0.y += p * v0.y; a0.z += p * v0.z; a0.w += p * v0.w;
    a1.x += p * v1.x; a1.y += p * v1.y; a1.z += p * v1.z; a1.w += p * v1.w;
    a2.x += p * v2.x; a2.y += p * v2.y; a2.z += p * v2.z; a2.w += p * v2.w;
    a3.x += p * v3.x; a3.y += p * v3.y; a3.z += p * v3.z; a3.w += p * v3.w;
  }
  float inv = 1.f / l;
  float* op = o + ((size_t)bt * J_DIM + q) * 128 + h * 16;
  st4(op,      float4{a0.x * inv, a0.y * inv, a0.z * inv, a0.w * inv});
  st4(op + 4,  float4{a1.x * inv, a1.y * inv, a1.z * inv, a1.w * inv});
  st4(op + 8,  float4{a2.x * inv, a2.y * inv, a2.z * inv, a2.w * inv});
  st4(op + 12, float4{a3.x * inv, a3.y * inv, a3.z * inv, a3.w * inv});
}

__global__ void attn_temporal_kernel(const float* __restrict__ qkv, float* __restrict__ o) {
  int idx = blockIdx.x * blockDim.x + threadIdx.x;
  if (idx >= B_DIM * J_DIM * N_HEADS * T_DIM) return;
  int qt = idx % T_DIM;
  int h  = (idx / T_DIM) % N_HEADS;
  int bj = idx / (T_DIM * N_HEADS);
  int b  = bj / J_DIM, j = bj % J_DIM;
  const float scale = 0.25f;
  const float* qp = qkv + (((size_t)(b * T_DIM + qt)) * J_DIM + j) * 384 + h * 16;
  float4 q0 = ld4(qp), q1 = ld4(qp + 4), q2 = ld4(qp + 8), q3 = ld4(qp + 12);
  float m = -1e30f, l = 0.f;
  float4 a0 = {}, a1 = {}, a2 = {}, a3 = {};
  for (int kt = 0; kt < T_DIM; ++kt) {   // online softmax over 243 keys
    const float* kp = qkv + (((size_t)(b * T_DIM + kt)) * J_DIM + j) * 384 + 128 + h * 16;
    float s = dot4(q0, ld4(kp)) + dot4(q1, ld4(kp + 4)) +
              dot4(q2, ld4(kp + 8)) + dot4(q3, ld4(kp + 12));
    s *= scale;
    float mn   = fmaxf(m, s);
    float corr = __expf(m - mn);
    float p    = __expf(s - mn);
    l = l * corr + p;
    const float* vp = kp + 128;
    float4 v0 = ld4(vp), v1 = ld4(vp + 4), v2 = ld4(vp + 8), v3 = ld4(vp + 12);
    a0.x = a0.x * corr + p * v0.x; a0.y = a0.y * corr + p * v0.y;
    a0.z = a0.z * corr + p * v0.z; a0.w = a0.w * corr + p * v0.w;
    a1.x = a1.x * corr + p * v1.x; a1.y = a1.y * corr + p * v1.y;
    a1.z = a1.z * corr + p * v1.z; a1.w = a1.w * corr + p * v1.w;
    a2.x = a2.x * corr + p * v2.x; a2.y = a2.y * corr + p * v2.y;
    a2.z = a2.z * corr + p * v2.z; a2.w = a2.w * corr + p * v2.w;
    a3.x = a3.x * corr + p * v3.x; a3.y = a3.y * corr + p * v3.y;
    a3.z = a3.z * corr + p * v3.z; a3.w = a3.w * corr + p * v3.w;
    m = mn;
  }
  float inv = 1.f / l;
  float* op = o + (((size_t)(b * T_DIM + qt)) * J_DIM + j) * 128 + h * 16;
  st4(op,      float4{a0.x * inv, a0.y * inv, a0.z * inv, a0.w * inv});
  st4(op + 4,  float4{a1.x * inv, a1.y * inv, a1.z * inv, a1.w * inv});
  st4(op + 8,  float4{a2.x * inv, a2.y * inv, a2.z * inv, a2.w * inv});
  st4(op + 12, float4{a3.x * inv, a3.y * inv, a3.z * inv, a3.w * inv});
}

// ------------------------------- Mamba2 parts -------------------------------
__global__ void conv_silu_kernel(const float* __restrict__ zx, int stride, int off,
                                 const float* __restrict__ w, const float* __restrict__ b,
                                 float* __restrict__ out, int ntok, int L) {
  int i = blockIdx.x * 256 + threadIdx.x;
  if (i >= ntok * 512) return;
  int tok = i >> 9, c = i & 511;
  int t = tok % L;
  float4 wc = ld4(w + c * 4);
  float acc = b[c];
  const float* col = zx + (size_t)(tok - t) * stride + off + c;  // seq start
  if (t >= 3) {
    acc += col[(size_t)(t - 3) * stride] * wc.x + col[(size_t)(t - 2) * stride] * wc.y +
           col[(size_t)(t - 1) * stride] * wc.z + col[(size_t)t * stride] * wc.w;
  } else {
#pragma unroll
    for (int k = 0; k < 4; ++k) {
      int tp = t + k - 3;
      float wk = (k == 0) ? wc.x : (k == 1) ? wc.y : (k == 2) ? wc.z : wc.w;
      if (tp >= 0) acc += col[(size_t)tp * stride] * wk;
    }
  }
  out[i] = silu_f(acc);
}

__global__ void dt_softplus_kernel(const float* __restrict__ zx, int stride, int dtoff,
                                   const float* __restrict__ dt_bias, float* __restrict__ dt,
                                   int ntok, int nhshift) {
  int i = blockIdx.x * 256 + threadIdx.x;
  if (i >= (ntok << nhshift)) return;
  int t = i >> nhshift, h = i & ((1 << nhshift) - 1);
  float v = zx[(size_t)t * stride + dtoff + h] + dt_bias[h];
  dt[i] = (v > 20.f) ? v : log1pf(__expf(v));
}

// Sequential SSM scan: one block per (seq, head); state S[hd,128] striped over
// waves (p rows) x lanes (4 n-values each). The next timestep's 2KB xBC row is
// async-prefetched global->LDS (ASYNCcnt) while the current step computes.
__global__ __launch_bounds__(128) void mamba_scan_kernel(
    const float* __restrict__ xbc, const float* __restrict__ dt,
    const float* __restrict__ A_log, const float* __restrict__ Dp,
    float* __restrict__ y, int nseq, int L, int nheads, int headdim) {
  __shared__ float sbuf[2][512];
  int blk  = blockIdx.x;
  int seq  = blk / nheads, h = blk % nheads;
  int tid  = threadIdx.x;
  int wave = tid >> 5, lane = tid & 31;
  int rpw  = headdim >> 2;   // rows of p per wave (4 waves)
  int p0   = wave * rpw;
  float A  = -__expf(A_log[h]);
  float Dh = Dp[h];
  const float* seqbase = xbc + (size_t)seq * L * 512;
  float4 S[8];
#pragma unroll
  for (int r = 0; r < 8; ++r) S[r] = float4{0.f, 0.f, 0.f, 0.f};
  // preload t = 0 (one b128 per thread covers the 512-float row)
  async_copy16(seqbase + tid * 4, &sbuf[0][tid * 4]);
  async_wait0();
  __syncthreads();
  for (int t = 0; t < L; ++t) {
    if (t + 1 < L)  // overlap: fetch next row while computing this one
      async_copy16(seqbase + (size_t)(t + 1) * 512 + tid * 4, &sbuf[(t + 1) & 1][tid * 4]);
    const float* row = sbuf[t & 1];
    size_t tok = (size_t)seq * L + t;
    float4 Bv = ld4(row + 256 + lane * 4);
    float4 Cv = ld4(row + 384 + lane * 4);
    float dtv = dt[tok * nheads + h];
    float dA  = __expf(dtv * A);
    float* yrow = y + tok * 256 + h * headdim;
    for (int r = 0; r < rpw; ++r) {
      float xv = row[h * headdim + p0 + r];
      float dx = dtv * xv;
      S[r].x = S[r].x * dA + dx * Bv.x;
      S[r].y = S[r].y * dA + dx * Bv.y;
      S[r].z = S[r].z * dA + dx * Bv.z;
      S[r].w = S[r].w * dA + dx * Bv.w;
      float part = wave_sum(dot4(S[r], Cv));
      if (lane == 0) yrow[p0 + r] = part + Dh * xv;
    }
    async_wait0();     // next-step row landed in LDS
    __syncthreads();   // visible to all waves before next iteration
  }
}

// gated RMSNorm: y = (y * silu(z)); y *= rsqrt(mean(y^2)+eps) * norm_w   (d_inner=256)
__global__ __launch_bounds__(128) void gate_rms_kernel(
    float* __restrict__ y, const float* __restrict__ zx, int zstride,
    const float* __restrict__ nw, int ntok) {
  int wv   = (blockIdx.x * blockDim.x + threadIdx.x) >> 5;
  int lane = threadIdx.x & 31;
  if (wv >= ntok) return;
  const int c = lane * 8;  // 256 ch / 32 lanes
  float4 z0 = ld4(zx + (size_t)wv * zstride + c);
  float4 z1 = ld4(zx + (size_t)wv * zstride + c + 4);
  float4 y0 = ld4(y + (size_t)wv * 256 + c);
  float4 y1 = ld4(y + (size_t)wv * 256 + c + 4);
  float4 g0 = { y0.x * silu_f(z0.x), y0.y * silu_f(z0.y), y0.z * silu_f(z0.z), y0.w * silu_f(z0.w) };
  float4 g1 = { y1.x * silu_f(z1.x), y1.y * silu_f(z1.y), y1.z * silu_f(z1.z), y1.w * silu_f(z1.w) };
  float q = wave_sum(dot4(g0, g0) + dot4(g1, g1));
  float rs = rsqrtf(q * (1.f / 256.f) + EPS_F);
  float4 w0 = ld4(nw + c), w1 = ld4(nw + c + 4);
  st4(y + (size_t)wv * 256 + c,     float4{g0.x * rs * w0.x, g0.y * rs * w0.y, g0.z * rs * w0.z, g0.w * rs * w0.w});
  st4(y + (size_t)wv * 256 + c + 4, float4{g1.x * rs * w1.x, g1.y * rs * w1.y, g1.z * rs * w1.z, g1.w * rs * w1.w});
}

// --------------------------- layout transforms ------------------------------
__global__ void t_fwd_kernel(const float* __restrict__ in, float* __restrict__ out) {
  int i = blockIdx.x * 256 + threadIdx.x;      // over NTOK*32 float4s
  if (i >= NTOK * 32) return;
  int c4 = (i & 31) * 4, tok = i >> 5;
  int j = tok % J_DIM, bt = tok / J_DIM;
  int t = bt % T_DIM, b = bt / T_DIM;
  st4(out + (((size_t)(b * J_DIM + j) * T_DIM + t) << 7) + c4, ld4(in + ((size_t)tok << 7) + c4));
}
__global__ void t_bwd_kernel(const float* __restrict__ in, float* __restrict__ out) {
  int i = blockIdx.x * 256 + threadIdx.x;
  if (i >= NTOK * 32) return;
  int c4 = (i & 31) * 4, tok = i >> 5;
  int t = tok % T_DIM, bj = tok / T_DIM;
  int j = bj % J_DIM, b = bj / J_DIM;
  st4(out + (((size_t)(b * T_DIM + t) * J_DIM + j) << 7) + c4, ld4(in + ((size_t)tok << 7) + c4));
}
__global__ void flip_ch_kernel(const float* __restrict__ in, float* __restrict__ out, int n4) {
  int i = blockIdx.x * 256 + threadIdx.x;
  if (i >= n4) return;
  int c4 = (i & 31) * 4;
  size_t t = (size_t)(i >> 5);
  float4 v = ld4(in + (t << 7) + c4);
  st4(out + (t << 7) + (124 - c4), float4{v.w, v.z, v.y, v.x});
}

// -------------------------------- fusion ------------------------------------
__global__ __launch_bounds__(128) void fusion_kernel(
    const float* __restrict__ xa, const float* __restrict__ xm,
    const float* __restrict__ fw, const float* __restrict__ fb,
    float* __restrict__ out, int ntok) {
  int wv   = (blockIdx.x * blockDim.x + threadIdx.x) >> 5;
  int lane = threadIdx.x & 31;
  if (wv >= ntok) return;
  const int c = lane * 4;
  float4 a = ld4(xa + (size_t)wv * 128 + c);
  float4 m = ld4(xm + (size_t)wv * 128 + c);
  float p0 = wave_sum(dot4(a, ld4(fw + c)) + dot4(m, ld4(fw + 128 + c)));
  float p1 = wave_sum(dot4(a, ld4(fw + 256 + c)) + dot4(m, ld4(fw + 384 + c)));
  float a0 = p0 + fb[0], a1 = p1 + fb[1];
  float mx = fmaxf(a0, a1);
  float e0 = __expf(a0 - mx), e1 = __expf(a1 - mx);
  float inv = 1.f / (e0 + e1);
  float w0 = e0 * inv, w1 = e1 * inv;
  st4(out + (size_t)wv * 128 + c,
      float4{a.x * w0 + m.x * w1, a.y * w0 + m.y * w1, a.z * w0 + m.z * w1, a.w * w0 + m.w * w1});
}

// ================================ host side =================================
// Input index map (setup_inputs dict order, recursively flattened):
// 0:x | att_s 1..13 | att_t 14..26 | mamba_s 27..34 + m 35..42
// | mamba_t 43..50 + m 51..58 + m_inv 59..66 | 67:fusion_w 68:fusion_b
extern "C" void kernel_launch(void* const* d_in, const int* in_sizes, int n_in,
                              void* d_out, int out_size, void* d_ws, size_t ws_size,
                              hipStream_t stream) {
  (void)in_sizes; (void)n_in; (void)out_size; (void)ws_size;
  auto F = [&](int i) { return (const float*)d_in[i]; };

  // ---- workspace carve-up (floats) ----
  float* base = (float*)d_ws;
  size_t off = 0;
  auto alloc = [&](size_t n) { float* p = base + off; off += n; return p; };
  float* bA   = alloc((size_t)NTOK * 128);  // x_att branch
  float* bB   = alloc((size_t)NTOK * 128);  // mamba branch
  float* bLN  = alloc((size_t)NTOK * 128);
  float* bBIG = alloc((size_t)NTOK * 784);  // qkv / zxbcdt / mlp hidden
  float* bMID = alloc((size_t)NTOK * 512);  // conv out / attn out
  float* bY   = alloc((size_t)NTOK * 256);  // mamba inner y
  float* bP   = alloc((size_t)NTOK * 128);  // proj / mlp / out_proj result
  float* bT   = alloc((size_t)NTOK * 128);  // temporal-transposed x
  float* bN   = alloc((size_t)NTOK * 128);  // ln(xt)
  float* bNF  = alloc((size_t)NTOK * 128);  // channel-flipped ln(xt)
  float* bDT  = alloc((size_t)NTOK * 16);
  _Float16* w16 = (_Float16*)(base + off);
  size_t w16off = 0;

  auto cvt = [&](int idx, int n) {
    _Float16* dst = w16 + w16off;
    w16off += (size_t)((n + 63) & ~63);
    f32_to_f16_kernel<<<dim3((n / 4 + 255) / 256), dim3(256), 0, stream>>>(F(idx), dst, n / 4);
    return (const _Float16*)dst;
  };
  // stage all GEMM weights as f16 once per launch (deterministic)
  const _Float16* w_qkv_s = cvt(5, 384 * 128);
  const _Float16* w_prj_s = cvt(6, 128 * 128);
  const _Float16* w_fc1_s = cvt(8, 512 * 128);
  const _Float16* w_fc2_s = cvt(10, 128 * 512);
  const _Float16* w_qkv_t = cvt(18, 384 * 128);
  const _Float16* w_prj_t = cvt(19, 128 * 128);
  const _Float16* w_fc1_t = cvt(21, 512 * 128);
  const _Float16* w_fc2_t = cvt(23, 128 * 512);
  const _Float16* w_msf1  = cvt(31, 512 * 128);
  const _Float16* w_msf2  = cvt(33, 128 * 512);
  const _Float16* w_msin  = cvt(35, 776 * 128);
  const _Float16* w_msout = cvt(42, 128 * 256);
  const _Float16* w_mtf1  = cvt(47, 512 * 128);
  const _Float16* w_mtf2  = cvt(49, 128 * 512);
  const _Float16* w_mtin  = cvt(51, 784 * 128);
  const _Float16* w_mtout = cvt(58, 128 * 256);
  const _Float16* w_miin  = cvt(59, 784 * 128);
  const _Float16* w_miout = cvt(66, 128 * 256);

  auto gemm = [&](int act, const float* A, const _Float16* W, const float* bias,
                  float* C, int M, int N, int K) {
    dim3 g((N + 63) / 64, (M + 63) / 64), blk(128);
    if (act == 0) gemm_wmma_f16<0><<<g, blk, 0, stream>>>(A, W, bias, C, M, N, K);
    else          gemm_wmma_f16<1><<<g, blk, 0, stream>>>(A, W, bias, C, M, N, K);
  };
  auto ln = [&](const float* x, const float* wv, const float* bv, float* y) {
    ln_kernel<<<dim3((NTOK + 3) / 4), dim3(128), 0, stream>>>(x, wv, bv, y, NTOK);
  };
  auto addw = [&](float* x, const float* y, const float* ls) {
    add_scaled_kernel<<<dim3((NTOK * 32 + 255) / 256), dim3(256), 0, stream>>>(x, y, ls, NTOK * 32);
  };

  auto mamba2 = [&](const float* xin, int mb, int hd, int L,
                    const _Float16* in16, const _Float16* out16, float* outb) {
    int nh = 256 / hd;           // 8 (hd=32) or 16 (hd=16)
    int nhs = (nh == 8) ? 3 : 4;
    int P  = 768 + nh;
    gemm(0, xin, in16, nullptr, bBIG, NTOK, P, 128);  // zxbcdt
    conv_silu_kernel<<<dim3((NTOK * 512 + 255) / 256), dim3(256), 0, stream>>>(
        bBIG, P, 256, F(mb + 1), F(mb + 2), bMID, NTOK, L);
    dt_softplus_kernel<<<dim3(((NTOK << nhs) + 255) / 256), dim3(256), 0, stream>>>(
        bBIG, P, 768, F(mb + 3), bDT, NTOK, nhs);
    mamba_scan_kernel<<<dim3((NTOK / L) * nh), dim3(128), 0, stream>>>(
        bMID, bDT, F(mb + 4), F(mb + 5), bY, NTOK / L, L, nh, hd);
    gate_rms_kernel<<<dim3((NTOK + 3) / 4), dim3(128), 0, stream>>>(bY, bBIG, P, F(mb + 6), NTOK);
    gemm(0, bY, out16, nullptr, outb, NTOK, 128, 256);
  };

  auto tblock = [&](float* x, int b0, int mode, const _Float16* q16, const _Float16* p16,
                    const _Float16* f116, const _Float16* f216) {
    ln(x, F(b0 + 0), F(b0 + 1), bLN);
    gemm(0, bLN, q16, nullptr, bBIG, NTOK, 384, 128);  // qkv
    int na = B_DIM * T_DIM * N_HEADS * J_DIM;
    if (mode == 0)
      attn_spatial_kernel<<<dim3((na + 127) / 128), dim3(128), 0, stream>>>(bBIG, bMID);
    else
      attn_temporal_kernel<<<dim3((na + 127) / 128), dim3(128), 0, stream>>>(bBIG, bMID);
    gemm(0, bMID, p16, F(b0 + 6), bP, NTOK, 128, 128);   // proj
    addw(x, bP, F(b0 + 11));                             // + ls1 * attn
    ln(x, F(b0 + 2), F(b0 + 3), bLN);
    gemm(1, bLN, f116, F(b0 + 8), bBIG, NTOK, 512, 128); // fc1 + gelu
    gemm(0, bBIG, f216, F(b0 + 10), bP, NTOK, 128, 512); // fc2
    addw(x, bP, F(b0 + 12));                             // + ls2 * mlp
  };

  // ---------------------------- pipeline ------------------------------------
  (void)hipMemcpyAsync(bA, F(0), (size_t)NTOK * 128 * sizeof(float), hipMemcpyDeviceToDevice, stream);
  (void)hipMemcpyAsync(bB, F(0), (size_t)NTOK * 128 * sizeof(float), hipMemcpyDeviceToDevice, stream);

  // attention branch
  tblock(bA, 1, 0, w_qkv_s, w_prj_s, w_fc1_s, w_fc2_s);   // spatial
  tblock(bA, 14, 1, w_qkv_t, w_prj_t, w_fc1_t, w_fc2_t);  // temporal

  // spatial mamba block (seqs of J=17, headdim 32)
  ln(bB, F(27), F(28), bLN);
  mamba2(bLN, 35, 32, 17, w_msin, w_msout, bP);
  addw(bB, bP, nullptr);
  ln(bB, F(29), F(30), bLN);
  gemm(1, bLN, w_msf1, F(32), bBIG, NTOK, 512, 128);
  gemm(0, bBIG, w_msf2, F(34), bP, NTOK, 128, 512);
  addw(bB, bP, nullptr);

  // temporal mamba block (seqs of T=243, headdim 16, fwd + channel-flipped)
  t_fwd_kernel<<<dim3((NTOK * 32 + 255) / 256), dim3(256), 0, stream>>>(bB, bT);
  ln(bT, F(43), F(44), bN);
  flip_ch_kernel<<<dim3((NTOK * 32 + 255) / 256), dim3(256), 0, stream>>>(bN, bNF, NTOK * 32);
  mamba2(bN, 51, 16, 243, w_mtin, w_mtout, bP);
  addw(bT, bP, nullptr);
  mamba2(bNF, 59, 16, 243, w_miin, w_miout, bP);
  addw(bT, bP, nullptr);
  ln(bT, F(45), F(46), bN);
  gemm(1, bN, w_mtf1, F(48), bBIG, NTOK, 512, 128);
  gemm(0, bBIG, w_mtf2, F(50), bP, NTOK, 128, 512);
  addw(bT, bP, nullptr);
  t_bwd_kernel<<<dim3((NTOK * 32 + 255) / 256), dim3(256), 0, stream>>>(bT, bB);

  // fusion -> d_out
  fusion_kernel<<<dim3((NTOK + 3) / 4), dim3(128), 0, stream>>>(
      bA, bB, F(67), F(68), (float*)d_out, NTOK);
}
